// Hawk_60790967108258
// MI455X (gfx1250) — compile-verified, matrix operand order checked by reading, not run on hardware
//
#include <hip/hip_runtime.h>

// ---------------- constants ----------------
#define BB 2
#define SS 4096
#define HH 2048
#define RR 2048
#define NBLK 16
#define BW 128
#define CW 4
#define MTOT (BB*SS)          // 8192 tokens
#define NCH 64                // scan chunks
#define CHL 64                // chunk length (NCH*CHL == SS)

typedef __attribute__((ext_vector_type(16))) __bf16 v16bf;
typedef __attribute__((ext_vector_type(8)))  __bf16 v8bf;
typedef __attribute__((ext_vector_type(8)))  float  v8f;
typedef __attribute__((ext_vector_type(4)))  int    v4i;

// ---------------- CDNA5 async global->LDS copy (16B), ASYNCcnt tracked ----------
#if defined(__gfx1250__) && __has_builtin(__builtin_amdgcn_global_load_async_to_lds_b128)
#define HAVE_ASYNC_LDS 1
#else
#define HAVE_ASYNC_LDS 0
#endif

typedef __attribute__((address_space(1))) v4i* gv4i_p;
typedef __attribute__((address_space(3))) v4i* lv4i_p;

__device__ __forceinline__ void async_copy16(const void* g, void* l) {
#if HAVE_ASYNC_LDS
  __builtin_amdgcn_global_load_async_to_lds_b128(
      (gv4i_p)const_cast<void*>(g), (lv4i_p)l, 0, 0);
#else
  *(uint4*)l = *(const uint4*)g;
#endif
}

__device__ __forceinline__ void wait_async() {
#if HAVE_ASYNC_LDS && __has_builtin(__builtin_amdgcn_s_wait_asynccnt)
  __builtin_amdgcn_s_wait_asynccnt(0);
#endif
}

// ---------------- fragment assembly from row-major LDS tile ----------------
// CDNA5 wmma 16x16x32 bf16: lane holds row (L%16); elems 0-7 = K kk+half*8+0..7,
// elems 8-15 = K kk+16+half*8+0..7.
__device__ __forceinline__ v16bf load_frag(const __bf16* tile, int row, int ldk,
                                           int kk, int half) {
  const v8bf* p0 = (const v8bf*)(tile + row * ldk + kk + half * 8);
  const v8bf* p1 = (const v8bf*)(tile + row * ldk + kk + 16 + half * 8);
  v8bf lo = *p0, hi = *p1;
  v16bf f;
#pragma unroll
  for (int i = 0; i < 8; ++i) { f[i] = lo[i]; f[8 + i] = hi[i]; }
  return f;
}

#define WMMA_BF16(a, b, c) \
  __builtin_amdgcn_wmma_f32_16x16x32_bf16(false, (a), false, (b), (short)0, (c), false, false)

// ---------------- casts ----------------
__global__ void cast_f32_bf16(const float* __restrict__ in, __bf16* __restrict__ out, int n) {
  for (int i = blockIdx.x * blockDim.x + threadIdx.x; i < n; i += gridDim.x * blockDim.x)
    out[i] = (__bf16)in[i];
}

// in: [NBLK][BW][BW] (i,j) -> out transposed (j,i), bf16
__global__ void transpose_cast_blocks(const float* __restrict__ in, __bf16* __restrict__ out, int n) {
  for (int i = blockIdx.x * blockDim.x + threadIdx.x; i < n; i += gridDim.x * blockDim.x) {
    int nb = i / (BW * BW);
    int rem = i % (BW * BW);
    int ri = rem / BW, cj = rem % BW;
    out[nb * BW * BW + cj * BW + ri] = (__bf16)in[i];
  }
}

// ---------------- generic bf16 WMMA GEMM: C[M,N] = A[M,K] * B[N,K]^T ----------------
// 128x128 block tile, K-stage 64, double-buffered async LDS, 256 threads (8 waves).
// Wave (wm = wid&3, wn = wid>>2) owns a 32(M)x64(N) sub-tile: 2 M-frags x 4 N-frags.
#define GBM 128
#define GBN 128
#define GBK 64

__global__ __launch_bounds__(256) void gemm_bf16(const __bf16* __restrict__ A,
                                                 const __bf16* __restrict__ Bw,
                                                 float* __restrict__ C,
                                                 int M, int N, int K) {
  __shared__ __align__(16) __bf16 As[2][GBM * GBK];
  __shared__ __align__(16) __bf16 Bs[2][GBN * GBK];
  const int tid = threadIdx.x;
  const int wid = tid >> 5, lane = tid & 31;
  const int half = lane >> 4, l16 = lane & 15;
  const int wm = wid & 3, wn = wid >> 2;
  const int mBase = blockIdx.y * GBM, nBase = blockIdx.x * GBN;
  const int lr = tid >> 1;            // 0..127 tile row
  const int lc = (tid & 1) * 32;      // 0/32 halfs (64B per thread per tile)
  v8f acc[2][4] = {};

  // prologue: stage 0
  {
    const __bf16* ga = A + (size_t)(mBase + lr) * K + lc;
    const __bf16* gb = Bw + (size_t)(nBase + lr) * K + lc;
    __bf16* la = &As[0][lr * GBK + lc];
    __bf16* lb = &Bs[0][lr * GBK + lc];
#pragma unroll
    for (int u = 0; u < 4; ++u) {
      async_copy16(ga + u * 8, la + u * 8);
      async_copy16(gb + u * 8, lb + u * 8);
    }
  }
  wait_async();
  __syncthreads();

  int cur = 0;
  for (int k0 = 0; k0 < K; k0 += GBK) {
    const int nxt = cur ^ 1;
    if (k0 + GBK < K) {
      const __bf16* ga = A + (size_t)(mBase + lr) * K + (k0 + GBK) + lc;
      const __bf16* gb = Bw + (size_t)(nBase + lr) * K + (k0 + GBK) + lc;
      __bf16* la = &As[nxt][lr * GBK + lc];
      __bf16* lb = &Bs[nxt][lr * GBK + lc];
#pragma unroll
      for (int u = 0; u < 4; ++u) {
        async_copy16(ga + u * 8, la + u * 8);
        async_copy16(gb + u * 8, lb + u * 8);
      }
    }
#pragma unroll
    for (int kk = 0; kk < GBK; kk += 32) {
      v16bf a0 = load_frag(&As[cur][0], (wm * 2 + 0) * 16 + l16, GBK, kk, half);
      v16bf a1 = load_frag(&As[cur][0], (wm * 2 + 1) * 16 + l16, GBK, kk, half);
#pragma unroll
      for (int nf = 0; nf < 4; ++nf) {
        v16bf bf = load_frag(&Bs[cur][0], (wn * 4 + nf) * 16 + l16, GBK, kk, half);
        acc[0][nf] = WMMA_BF16(a0, bf, acc[0][nf]);
        acc[1][nf] = WMMA_BF16(a1, bf, acc[1][nf]);
      }
    }
    wait_async();
    __syncthreads();
    cur = nxt;
  }

  const int col0 = nBase + wn * 64 + l16;
#pragma unroll
  for (int mf = 0; mf < 2; ++mf) {
    const int row = mBase + (wm * 2 + mf) * 16 + 8 * half;
#pragma unroll
    for (int nf = 0; nf < 4; ++nf) {
#pragma unroll
      for (int r = 0; r < 8; ++r)
        C[(size_t)(row + r) * N + col0 + nf * 16] = acc[mf][nf][r];
    }
  }
}

// ---------------- causal depthwise conv ----------------
// xr lives in xy[:, 0:RR] (row stride 2*RR). Writes xc fp32 + bf16 copy.
__global__ void conv_kernel(const float* __restrict__ xy, const float* __restrict__ conv_w,
                            const float* __restrict__ conv_b,
                            float* __restrict__ xc_f, __bf16* __restrict__ xc_bf, int n) {
  for (int i = blockIdx.x * blockDim.x + threadIdx.x; i < n; i += gridDim.x * blockDim.x) {
    int r = i % RR;
    int t = (i / RR) % SS;
    int b = i / (RR * SS);
    float acc = conv_b[r];
#pragma unroll
    for (int k = 0; k < CW; ++k) {
      int tt = t - (CW - 1) + k;
      if (tt >= 0)
        acc += conv_w[k * RR + r] * xy[(size_t)(b * SS + tt) * (2 * RR) + r];
    }
    xc_f[i] = acc;
    xc_bf[i] = (__bf16)acc;
  }
}

// ---------------- fused block-diag gates + RG-LRU pointwise ----------------
// grid: (NBLK, MTOT/64). Computes a = exp(log_a) and norm_x for the scan.
__global__ __launch_bounds__(256) void gates_kernel(
    const __bf16* __restrict__ xc_bf, const float* __restrict__ xc_f,
    const __bf16* __restrict__ igT, const __bf16* __restrict__ agT,
    const float* __restrict__ ig_b, const float* __restrict__ ag_b,
    const float* __restrict__ a_param,
    float* __restrict__ a_out, float* __restrict__ nx_out) {
  const int nblk = blockIdx.x;
  const int tokBase = blockIdx.y * 64;
  __shared__ __align__(16) __bf16 As[64 * 32];
  __shared__ __align__(16) __bf16 BsI[128 * 32];
  __shared__ __align__(16) __bf16 BsA[128 * 32];
  const int tid = threadIdx.x, wid = tid >> 5, lane = tid & 31;
  const int half = lane >> 4, l16 = lane & 15;
  const int msub = wid & 3, npair = wid >> 2;   // 4 M-subtiles x 2 N-halves
  v8f accx[4] = {}, acca[4] = {};
  const int lr = tid >> 2, lc = (tid & 3) << 3;       // A tile: 8 halfs/thread
  const int br = tid >> 1, bc = (tid & 1) << 4;       // B tiles: 16 halfs/thread
  for (int k0 = 0; k0 < BW; k0 += 32) {
    async_copy16(xc_bf + (size_t)(tokBase + lr) * RR + nblk * BW + k0 + lc,
                 As + lr * 32 + lc);
    const size_t wb = (size_t)nblk * BW * BW + (size_t)br * BW + k0 + bc;
    async_copy16(igT + wb,     BsI + br * 32 + bc);
    async_copy16(igT + wb + 8, BsI + br * 32 + bc + 8);
    async_copy16(agT + wb,     BsA + br * 32 + bc);
    async_copy16(agT + wb + 8, BsA + br * 32 + bc + 8);
    wait_async();
    __syncthreads();
    v16bf af = load_frag(As, msub * 16 + l16, 32, 0, half);
#pragma unroll
    for (int nf = 0; nf < 4; ++nf) {
      v16bf bi = load_frag(BsI, (npair * 4 + nf) * 16 + l16, 32, 0, half);
      accx[nf] = WMMA_BF16(af, bi, accx[nf]);
      v16bf ba = load_frag(BsA, (npair * 4 + nf) * 16 + l16, 32, 0, half);
      acca[nf] = WMMA_BF16(af, ba, acca[nf]);
    }
    __syncthreads();
  }
  const int token0 = tokBase + msub * 16 + 8 * half;
#pragma unroll
  for (int nf = 0; nf < 4; ++nf) {
    const int j = npair * 64 + nf * 16 + l16;
    const int rg = nblk * BW + j;
    const float bi = ig_b[rg], ba = ag_b[rg];
    float p = a_param[rg];
    const float sp = (p > 20.f) ? p : log1pf(__expf(p));   // softplus
#pragma unroll
    for (int r = 0; r < 8; ++r) {
      const int token = token0 + r;
      const float xgp = accx[nf][r] + bi;
      const float agp = acca[nf][r] + ba;
      const float gx = 1.f / (1.f + __expf(-xgp));
      const float ga = 1.f / (1.f + __expf(-agp));
      const float log_a = -8.f * ga * sp;
      const float aa = __expf(log_a);
      const float mult = sqrtf(fmaxf(-expm1f(2.f * log_a), 0.f));
      const size_t idx = (size_t)token * RR + rg;
      a_out[idx] = aa;
      nx_out[idx] = xc_f[idx] * gx * mult;
    }
  }
}

// ---------------- chunked linear-recurrence scan ----------------
__global__ void scan_chunk(float* __restrict__ a_arr, float* __restrict__ nx_arr,
                           float* __restrict__ prodA, float* __restrict__ hend) {
  const int tid = blockIdx.x * blockDim.x + threadIdx.x;  // BB*RR*NCH threads
  if (tid >= BB * RR * NCH) return;
  const int r = tid % RR;
  const int tmp = tid / RR;
  const int b = tmp % BB;
  const int chunk = tmp / BB;
  float pA = 1.f, h = 0.f;
  for (int i = 0; i < CHL; ++i) {
    const int t = chunk * CHL + i;
    const size_t idx = (size_t)(b * SS + t) * RR + r;
    const float a = a_arr[idx], nx = nx_arr[idx];
    pA *= a;
    h = a * h + nx;
    a_arr[idx] = pA;
    nx_arr[idx] = h;
  }
  const size_t sidx = ((size_t)chunk * BB + b) * RR + r;
  prodA[sidx] = pA;
  hend[sidx] = h;
}

__global__ void scan_carry(const float* __restrict__ prodA, const float* __restrict__ hend,
                           float* __restrict__ carry) {
  const int tid = blockIdx.x * blockDim.x + threadIdx.x;  // BB*RR threads
  if (tid >= BB * RR) return;
  const int r = tid % RR, b = tid / RR;
  float c = 0.f;
  for (int ch = 0; ch < NCH; ++ch) {
    const size_t sidx = ((size_t)ch * BB + b) * RR + r;
    carry[sidx] = c;
    c = prodA[sidx] * c + hend[sidx];
  }
}

// Pass C fused with gelu(y)*h -> bf16 operand for GEMM2.
__global__ void gelu_mul(const float* __restrict__ xy, const float* __restrict__ nx_arr,
                         const float* __restrict__ a_arr, const float* __restrict__ carry,
                         __bf16* __restrict__ g, int n) {
  for (int i = blockIdx.x * blockDim.x + threadIdx.x; i < n; i += gridDim.x * blockDim.x) {
    const int r = i % RR;
    const int t = (i / RR) % SS;
    const int b = i / (RR * SS);
    const int token = b * SS + t;
    const float y = xy[(size_t)token * (2 * RR) + RR + r];
    const int ch = t / CHL;
    const float c = carry[((size_t)ch * BB + b) * RR + r];
    const float h = nx_arr[i] + c * a_arr[i];
    const float gel = 0.5f * y * (1.f + erff(y * 0.70710678118654752f));
    g[i] = (__bf16)(gel * h);
  }
}

// ---------------- launch ----------------
extern "C" void kernel_launch(void* const* d_in, const int* in_sizes, int n_in,
                              void* d_out, int out_size, void* d_ws, size_t ws_size,
                              hipStream_t stream) {
  const float* x       = (const float*)d_in[0];  // [B,S,H]
  const float* W_xy    = (const float*)d_in[1];  // [2R,H]
  const float* ig_w    = (const float*)d_in[2];  // [NB,BW,BW]
  const float* ig_b    = (const float*)d_in[3];  // [NB,BW] flat
  const float* ag_w    = (const float*)d_in[4];
  const float* ag_b    = (const float*)d_in[5];
  const float* a_param = (const float*)d_in[6];  // [R]
  const float* conv_w  = (const float*)d_in[7];  // [CW,R]
  const float* conv_b  = (const float*)d_in[8];  // [R]
  const float* W_resid = (const float*)d_in[9];  // [H,R]
  float* out = (float*)d_out;

  char* ws = (char*)d_ws;
  size_t off = 0;
  auto alloc = [&](size_t bytes) { char* p = ws + off; off += (bytes + 255) & ~(size_t)255; return p; };
  __bf16* x_bf    = (__bf16*)alloc((size_t)MTOT * HH * 2);
  __bf16* Wxy_bf  = (__bf16*)alloc((size_t)2 * RR * HH * 2);
  __bf16* Wres_bf = (__bf16*)alloc((size_t)HH * RR * 2);
  __bf16* igT_bf  = (__bf16*)alloc((size_t)NBLK * BW * BW * 2);
  __bf16* agT_bf  = (__bf16*)alloc((size_t)NBLK * BW * BW * 2);
  float*  xy_f    = (float*)alloc((size_t)MTOT * 2 * RR * 4);
  float*  xc_f    = (float*)alloc((size_t)MTOT * RR * 4);
  __bf16* xc_bf   = (__bf16*)alloc((size_t)MTOT * RR * 2);
  float*  a_arr   = (float*)alloc((size_t)MTOT * RR * 4);
  float*  nx_arr  = (float*)alloc((size_t)MTOT * RR * 4);
  float*  prodA   = (float*)alloc((size_t)NCH * BB * RR * 4);
  float*  hendA   = (float*)alloc((size_t)NCH * BB * RR * 4);
  float*  carry   = (float*)alloc((size_t)NCH * BB * RR * 4);
  __bf16* g_bf    = (__bf16*)alloc((size_t)MTOT * RR * 2);
  (void)ws_size;

  const int TPB = 256;
  {
    int n = MTOT * HH;
    cast_f32_bf16<<<dim3((n + TPB - 1) / TPB), dim3(TPB), 0, stream>>>(x, x_bf, n);
    n = 2 * RR * HH;
    cast_f32_bf16<<<dim3((n + TPB - 1) / TPB), dim3(TPB), 0, stream>>>(W_xy, Wxy_bf, n);
    n = HH * RR;
    cast_f32_bf16<<<dim3((n + TPB - 1) / TPB), dim3(TPB), 0, stream>>>(W_resid, Wres_bf, n);
    n = NBLK * BW * BW;
    transpose_cast_blocks<<<dim3((n + TPB - 1) / TPB), dim3(TPB), 0, stream>>>(ig_w, igT_bf, n);
    transpose_cast_blocks<<<dim3((n + TPB - 1) / TPB), dim3(TPB), 0, stream>>>(ag_w, agT_bf, n);
  }
  // GEMM1: xy[8192,4096] = x_bf[8192,2048] * Wxy_bf[4096,2048]^T
  gemm_bf16<<<dim3((2 * RR) / GBN, MTOT / GBM), dim3(TPB), 0, stream>>>(
      x_bf, Wxy_bf, xy_f, MTOT, 2 * RR, HH);
  {
    int n = MTOT * RR;
    conv_kernel<<<dim3((n + TPB - 1) / TPB), dim3(TPB), 0, stream>>>(
        xy_f, conv_w, conv_b, xc_f, xc_bf, n);
  }
  gates_kernel<<<dim3(NBLK, MTOT / 64), dim3(TPB), 0, stream>>>(
      xc_bf, xc_f, igT_bf, agT_bf, ig_b, ag_b, a_param, a_arr, nx_arr);
  {
    int n = BB * RR * NCH;
    scan_chunk<<<dim3((n + TPB - 1) / TPB), dim3(TPB), 0, stream>>>(a_arr, nx_arr, prodA, hendA);
    n = BB * RR;
    scan_carry<<<dim3((n + TPB - 1) / TPB), dim3(TPB), 0, stream>>>(prodA, hendA, carry);
  }
  {
    int n = MTOT * RR;
    gelu_mul<<<dim3((n + TPB - 1) / TPB), dim3(TPB), 0, stream>>>(
        xy_f, nx_arr, a_arr, carry, g_bf, n);
  }
  // GEMM2: out[8192,2048] = g_bf[8192,2048] * Wres_bf[2048,2048]^T
  gemm_bf16<<<dim3(HH / GBN, MTOT / GBM), dim3(TPB), 0, stream>>>(
      g_bf, Wres_bf, out, MTOT, HH, RR);
  (void)in_sizes; (void)n_in; (void)out_size;
}
